// MaskedSelfAttention_47390669144297
// MI455X (gfx1250) — matrix operand order repaired and checked by reading, not compiled, not run on hardware
//
#include <hip/hip_runtime.h>
#include <hip/hip_bf16.h>
#include <math.h>

#define EMBED 2048
#define NHEAD 16
#define HDIM  128
#define BATCH 4
#define SEQ   2048
#define MROWS (BATCH * SEQ)   // 8192

#define KSTEP 64
#define LDS_STRIDE 72         // padded row stride (elements) for GEMM LDS tiles
#define KROW_STRIDE 136       // padded row stride (elements) for attention K tiles

typedef __attribute__((ext_vector_type(16))) __bf16 v16bf;
typedef __attribute__((ext_vector_type(8)))  __bf16 v8bf;
typedef __attribute__((ext_vector_type(4)))  __bf16 v4bf;
typedef __attribute__((ext_vector_type(8)))  float  v8f;

__device__ __forceinline__ v16bf cat8(v8bf lo, v8bf hi) {
  return __builtin_shufflevector(lo, hi, 0,1,2,3,4,5,6,7,8,9,10,11,12,13,14,15);
}

#define WMMA_BF16(A_, B_, C_) \
  __builtin_amdgcn_wmma_f32_16x16x32_bf16(false, (A_), false, (B_), (short)0, (C_), false, false)

// ---------------------------------------------------------------------------
// Async global -> LDS copy (CDNA5 GLOBAL_LOAD_ASYNC_TO_LDS_B128, ASYNCcnt).
// Prototype (probe-confirmed round 2): (int4 AS1*, int4 AS3*, Imm, Imm).
// ---------------------------------------------------------------------------
#if __has_builtin(__builtin_amdgcn_global_load_async_to_lds_b128)
#define HAS_ASYNC_LDS 1
#else
#define HAS_ASYNC_LDS 0
#endif

typedef int vec4i __attribute__((vector_size(16)));
typedef __attribute__((address_space(1))) vec4i g_vec4i;
typedef __attribute__((address_space(3))) vec4i l_vec4i;

__device__ __forceinline__ void async_b128(const __bf16* g, __bf16* l) {
#if HAS_ASYNC_LDS
  __builtin_amdgcn_global_load_async_to_lds_b128((g_vec4i*)g, (l_vec4i*)l, 0, 0);
#else
  *(v8bf*)l = *(const v8bf*)g;
#endif
}

#if HAS_ASYNC_LDS
#define WAIT_ASYNC(imm) asm volatile("s_wait_asynccnt " #imm ::: "memory")
#else
#define WAIT_ASYNC(imm) do {} while (0)
#endif

// ---------------------------------------------------------------------------
// FP32 -> BF16 conversion (vectorized x4; all sizes here are multiples of 4)
// ---------------------------------------------------------------------------
__global__ void cvt_f32_bf16(const float* __restrict__ in,
                             __bf16* __restrict__ out, int n4) {
  int i = blockIdx.x * blockDim.x + threadIdx.x;
  if (i < n4) {
    const float4 f = *(const float4*)(in + (size_t)i * 4);
    v4bf o;
    o[0] = (__bf16)f.x; o[1] = (__bf16)f.y;
    o[2] = (__bf16)f.z; o[3] = (__bf16)f.w;
    *(v4bf*)(out + (size_t)i * 4) = o;
  }
}

// ---------------------------------------------------------------------------
// C[M,N] = A[M,K] (bf16) @ W[N,K]^T (bf16) + bias
// Block = 4 waves -> 64x64 tile. K-steps of 64 staged into double-buffered
// LDS via async global->LDS copies; fragments read from LDS (ds_load_b128).
// ---------------------------------------------------------------------------
template<bool OUT_BF16>
__global__ __launch_bounds__(128) void gemm_wmma_bf16(
    const __bf16* __restrict__ A,
    const __bf16* __restrict__ W,
    const float*  __restrict__ bias,
    void* __restrict__ Cout,
    int M, int N, int K)
{
  __shared__ __bf16 sA[2][64 * LDS_STRIDE];
  __shared__ __bf16 sW[2][64 * LDS_STRIDE];

  const int tid    = threadIdx.x;
  const int lane   = tid & 31;
  const int wid    = tid >> 5;
  const int lm     = lane & 15;
  const int hiHalf = lane >> 4;
  const int m0 = blockIdx.y * 64;
  const int n0 = blockIdx.x * 64;
  const int wm = (wid >> 1) * 32;
  const int wn = (wid & 1) * 32;

  const __bf16* gA = A + (size_t)m0 * K;
  const __bf16* gW = W + (size_t)n0 * K;

  // Stage one 64x64 A tile + 64x64 W tile: 512 chunks of 16B each per matrix,
  // 4 chunks per thread per matrix. chunk c -> row c>>3, col (c&7)*8.
  auto stage = [&](int buf, int k0) {
#pragma unroll
    for (int i = 0; i < 4; ++i) {
      const int c   = tid + i * 128;
      const int row = c >> 3;
      const int col = (c & 7) * 8;
      async_b128(gA + (size_t)row * K + k0 + col, &sA[buf][row * LDS_STRIDE + col]);
      async_b128(gW + (size_t)row * K + k0 + col, &sW[buf][row * LDS_STRIDE + col]);
    }
  };

  v8f acc00 = {}, acc01 = {}, acc10 = {}, acc11 = {};

  stage(0, 0);
  const int nk = K / KSTEP;
  for (int kt = 0; kt < nk; ++kt) {
    WAIT_ASYNC(0x0);     // this wave's copies for buffer kt&1 landed in LDS
    __syncthreads();     // all waves' copies done AND all done reading kt-1&1
    if (kt + 1 < nk) stage((kt + 1) & 1, (kt + 1) * KSTEP);

    const __bf16* tA = sA[kt & 1];
    const __bf16* tW = sW[kt & 1];
#pragma unroll
    for (int kk = 0; kk < 2; ++kk) {
      const int ko = kk * 32;
      const __bf16* pa0 = tA + (wm + lm) * LDS_STRIDE + ko + hiHalf * 8;
      const __bf16* pa1 = pa0 + 16 * LDS_STRIDE;
      const __bf16* pw0 = tW + (wn + lm) * LDS_STRIDE + ko + hiHalf * 8;
      const __bf16* pw1 = pw0 + 16 * LDS_STRIDE;
      v16bf af0 = cat8(*(const v8bf*)pa0, *(const v8bf*)(pa0 + 16));
      v16bf af1 = cat8(*(const v8bf*)pa1, *(const v8bf*)(pa1 + 16));
      v16bf bf0 = cat8(*(const v8bf*)pw0, *(const v8bf*)(pw0 + 16));
      v16bf bf1 = cat8(*(const v8bf*)pw1, *(const v8bf*)(pw1 + 16));
      acc00 = WMMA_BF16(af0, bf0, acc00);
      acc01 = WMMA_BF16(af0, bf1, acc01);
      acc10 = WMMA_BF16(af1, bf0, acc10);
      acc11 = WMMA_BF16(af1, bf1, acc11);
    }
  }

  const int mrow = hiHalf * 8;
  const float b0 = bias[n0 + wn + lm];
  const float b1 = bias[n0 + wn + 16 + lm];
#pragma unroll
  for (int v = 0; v < 8; ++v) {
    const size_t r0 = (size_t)(m0 + wm + mrow + v) * N;
    const size_t r1 = (size_t)(m0 + wm + 16 + mrow + v) * N;
    const float c00 = acc00[v] + b0, c01 = acc01[v] + b1;
    const float c10 = acc10[v] + b0, c11 = acc11[v] + b1;
    if constexpr (OUT_BF16) {
      __bf16* C = (__bf16*)Cout;
      C[r0 + n0 + wn + lm]      = (__bf16)c00;
      C[r0 + n0 + wn + 16 + lm] = (__bf16)c01;
      C[r1 + n0 + wn + lm]      = (__bf16)c10;
      C[r1 + n0 + wn + 16 + lm] = (__bf16)c11;
    } else {
      float* C = (float*)Cout;
      C[r0 + n0 + wn + lm]      = c00;
      C[r0 + n0 + wn + 16 + lm] = c01;
      C[r1 + n0 + wn + lm]      = c10;
      C[r1 + n0 + wn + 16 + lm] = c11;
    }
  }
}

// ---------------------------------------------------------------------------
// Fused causal flash attention. Q/K/V/O layout: [B, T, H*D] bf16.
// One wave handles 16 query rows of one (b,h); streams key tiles of 32.
// K tiles are async-copied into per-wave double-buffered LDS: prefetch tile
// kt+1 (16 async ops), then s_wait_asynccnt 16 -> tile kt is ready while
// kt+1 stays in flight (ASYNCcnt completes in order). No barriers needed.
// ---------------------------------------------------------------------------
__global__ __launch_bounds__(128) void attn_flash_bf16(
    const __bf16* __restrict__ Q,
    const __bf16* __restrict__ K,
    const __bf16* __restrict__ V,
    __bf16* __restrict__ O)
{
  __shared__ __bf16 sK[4][2][32 * KROW_STRIDE]; // K tiles, double-buffered
  __shared__ __bf16 sV[4][HDIM * 32];           // V tile transposed: [d][key]
  __shared__ __bf16 sP[4][16 * 32];             // P tile row-major: [m][key]

  const int lane   = threadIdx.x & 31;
  const int w      = threadIdx.x >> 5;
  const int lm     = lane & 15;
  const int hiHalf = lane >> 4;
  const int b      = blockIdx.y >> 4;
  const int h      = blockIdx.y & 15;
  const int q0     = (blockIdx.x * 4 + w) * 16;

  const size_t headOff = (size_t)(b * SEQ) * EMBED + (size_t)h * HDIM;
  const __bf16* Qb = Q + headOff;
  const __bf16* Kb = K + headOff;
  const __bf16* Vb = V + headOff;
  __bf16*       Ob = O + headOff;

  __bf16* sKw = &sK[w][0][0];
  __bf16* sVw = &sV[w][0];
  __bf16* sPw = &sP[w][0];

  // Async-stage one 32x128 K tile: 512 chunks of 8 elems, 16 per lane.
  auto stageK = [&](int buf, int j0) {
    const __bf16* src = Kb + (size_t)j0 * EMBED;
    __bf16* dst = sKw + buf * (32 * KROW_STRIDE);
#pragma unroll
    for (int i = 0; i < 16; ++i) {
      const int c   = lane + i * 32;
      const int row = c >> 4;
      const int col = (c & 15) * 8;
      async_b128(src + (size_t)row * EMBED + col, dst + row * KROW_STRIDE + col);
    }
  };

  // Preload Q fragments (16 rows x 128 K = 4 fragments)
  v16bf qf[4];
  {
    const __bf16* qp = Qb + (size_t)(q0 + lm) * EMBED;
    const int ka = hiHalf * 8;
#pragma unroll
    for (int kk = 0; kk < 4; ++kk) {
      v8bf lo = *(const v8bf*)(qp + kk * 32 + ka);
      v8bf hi = *(const v8bf*)(qp + kk * 32 + ka + 16);
      qf[kk] = cat8(lo, hi);
    }
  }

  v8f acc[8] = {};
  float rowmax[8], rowsum[8];
#pragma unroll
  for (int v = 0; v < 8; ++v) { rowmax[v] = -INFINITY; rowsum[v] = 0.0f; }

  const int cmBase = hiHalf * 8;
  const float scale = 0.08838834764831845f;   // 1/sqrt(128)
  const float LOG2E = 1.4426950408889634f;
  const int nTiles = (q0 + 47) >> 5;          // key tiles with j0 <= q0+15

  stageK(0, 0);

  for (int kt = 0; kt < nTiles; ++kt) {
    const int j0 = kt * 32;

    // Prefetch next K tile, then wait for the current one (in-order ASYNCcnt)
    if (kt + 1 < nTiles) {
      stageK((kt + 1) & 1, j0 + 32);
      WAIT_ASYNC(0x10);
    } else {
      WAIT_ASYNC(0x0);
    }

    // Stage V tile transposed into LDS: sVw[d*32 + r] = V[j0+r][d]
    {
      const __bf16* vp0 = Vb + (size_t)j0 * EMBED + lane * 4;
#pragma unroll 4
      for (int r = 0; r < 32; ++r) {
        v4bf vv = *(const v4bf*)(vp0 + (size_t)r * EMBED);
#pragma unroll
        for (int i = 0; i < 4; ++i)
          sVw[(lane * 4 + i) * 32 + r] = vv[i];
      }
    }

    // S = Q K^T from the staged LDS K tile (ds_load_b128 fragments)
    v8f s0 = {}, s1 = {};
    {
      const __bf16* tK = sKw + (kt & 1) * (32 * KROW_STRIDE);
      const int kb2 = hiHalf * 8;
#pragma unroll
      for (int kk = 0; kk < 4; ++kk) {
        const __bf16* kp0 = tK + lm * KROW_STRIDE + kk * 32 + kb2;
        v16bf bf0 = cat8(*(const v8bf*)kp0, *(const v8bf*)(kp0 + 16));
        s0 = WMMA_BF16(qf[kk], bf0, s0);
        const __bf16* kp1 = tK + (16 + lm) * KROW_STRIDE + kk * 32 + kb2;
        v16bf bf1 = cat8(*(const v8bf*)kp1, *(const v8bf*)(kp1 + 16));
        s1 = WMMA_BF16(qf[kk], bf1, s1);
      }
    }

    // Scale + causal mask + online softmax (rows live across a 16-lane half)
    float p0[8], p1[8], alpha[8];
    const int n0g = j0 + lm, n1g = j0 + 16 + lm;
#pragma unroll
    for (int v = 0; v < 8; ++v) {
      const int qg = q0 + cmBase + v;
      float a = s0[v] * scale; if (n0g > qg) a = -1e30f;
      float c = s1[v] * scale; if (n1g > qg) c = -1e30f;
      p0[v] = a; p1[v] = c;
    }
#pragma unroll
    for (int v = 0; v < 8; ++v) {
      float x = fmaxf(p0[v], p1[v]);
      x = fmaxf(x, __shfl_xor(x, 1, 32));
      x = fmaxf(x, __shfl_xor(x, 2, 32));
      x = fmaxf(x, __shfl_xor(x, 4, 32));
      x = fmaxf(x, __shfl_xor(x, 8, 32));
      const float mnew = fmaxf(rowmax[v], x);
      alpha[v] = exp2f((rowmax[v] - mnew) * LOG2E);
      rowmax[v] = mnew;
      p0[v] = exp2f((p0[v] - mnew) * LOG2E);
      p1[v] = exp2f((p1[v] - mnew) * LOG2E);
      float sm = p0[v] + p1[v];
      sm += __shfl_xor(sm, 1, 32);
      sm += __shfl_xor(sm, 2, 32);
      sm += __shfl_xor(sm, 4, 32);
      sm += __shfl_xor(sm, 8, 32);
      rowsum[v] = rowsum[v] * alpha[v] + sm;
    }
#pragma unroll
    for (int nt = 0; nt < 8; ++nt)
#pragma unroll
      for (int v = 0; v < 8; ++v) acc[nt][v] *= alpha[v];

    // Store P (C layout -> row-major LDS) then reload as A fragment
#pragma unroll
    for (int v = 0; v < 8; ++v) {
      const int m = cmBase + v;
      sPw[m * 32 + lm]      = (__bf16)p0[v];
      sPw[m * 32 + 16 + lm] = (__bf16)p1[v];
    }
    v16bf pa;
    {
      const __bf16* pp = sPw + lm * 32 + hiHalf * 8;
      pa = cat8(*(const v8bf*)pp, *(const v8bf*)(pp + 16));
    }

    // acc += P (16x32) x V (32x128): B fragments contiguous from transposed LDS
    {
      const int kb2 = hiHalf * 8;
#pragma unroll
      for (int nt = 0; nt < 8; ++nt) {
        const __bf16* vpp = sVw + (16 * nt + lm) * 32 + kb2;
        v16bf vbf = cat8(*(const v8bf*)vpp, *(const v8bf*)(vpp + 16));
        acc[nt] = WMMA_BF16(pa, vbf, acc[nt]);
      }
    }
  }

  // Normalize and store O (bf16, token-major so Wo GEMM reads it directly)
#pragma unroll
  for (int v = 0; v < 8; ++v) {
    const float inv = 1.0f / rowsum[v];
    const int m = cmBase + v;
    __bf16* op = Ob + (size_t)(q0 + m) * EMBED;
#pragma unroll
    for (int nt = 0; nt < 8; ++nt)
      op[16 * nt + lm] = (__bf16)(acc[nt][v] * inv);
  }
}

// ---------------------------------------------------------------------------
extern "C" void kernel_launch(void* const* d_in, const int* in_sizes, int n_in,
                              void* d_out, int out_size, void* d_ws, size_t ws_size,
                              hipStream_t stream) {
  const float* x  = (const float*)d_in[0];
  const float* Wq = (const float*)d_in[1];
  const float* bq = (const float*)d_in[2];
  const float* Wk = (const float*)d_in[3];
  const float* bk = (const float*)d_in[4];
  const float* Wv = (const float*)d_in[5];
  const float* bv = (const float*)d_in[6];
  const float* Wo = (const float*)d_in[7];
  const float* bo = (const float*)d_in[8];

  const size_t SZ_X = (size_t)MROWS * EMBED;   // 16,777,216
  const size_t SZ_W = (size_t)EMBED * EMBED;   //  4,194,304

  __bf16* xb  = (__bf16*)d_ws;
  __bf16* wqb = xb  + SZ_X;
  __bf16* wkb = wqb + SZ_W;
  __bf16* wvb = wkb + SZ_W;
  __bf16* wob = wvb + SZ_W;
  __bf16* qb  = wob + SZ_W;
  __bf16* kb  = qb  + SZ_X;
  __bf16* vb  = kb  + SZ_X;
  __bf16* ob  = vb  + SZ_X;

  // 1) Convert activations + weights to bf16
  {
    int n4 = (int)(SZ_X / 4);
    cvt_f32_bf16<<<(n4 + 255) / 256, 256, 0, stream>>>(x, xb, n4);
    int w4 = (int)(SZ_W / 4);
    cvt_f32_bf16<<<(w4 + 255) / 256, 256, 0, stream>>>(Wq, wqb, w4);
    cvt_f32_bf16<<<(w4 + 255) / 256, 256, 0, stream>>>(Wk, wkb, w4);
    cvt_f32_bf16<<<(w4 + 255) / 256, 256, 0, stream>>>(Wv, wvb, w4);
    cvt_f32_bf16<<<(w4 + 255) / 256, 256, 0, stream>>>(Wo, wob, w4);
  }

  // 2) QKV projections (bf16 out)
  {
    dim3 grid(EMBED / 64, MROWS / 64);   // (32, 128)
    dim3 block(128);
    gemm_wmma_bf16<true><<<grid, block, 0, stream>>>(xb, wqb, bq, qb, MROWS, EMBED, EMBED);
    gemm_wmma_bf16<true><<<grid, block, 0, stream>>>(xb, wkb, bk, kb, MROWS, EMBED, EMBED);
    gemm_wmma_bf16<true><<<grid, block, 0, stream>>>(xb, wvb, bv, vb, MROWS, EMBED, EMBED);
  }

  // 3) Fused causal flash attention
  {
    dim3 grid(SEQ / (16 * 4), BATCH * NHEAD);  // (32, 64)
    dim3 block(128);
    attn_flash_bf16<<<grid, block, 0, stream>>>(qb, kb, vb, ob);
  }

  // 4) Output projection (f32 out)
  {
    dim3 grid(EMBED / 64, MROWS / 64);
    dim3 block(128);
    gemm_wmma_bf16<false><<<grid, block, 0, stream>>>(ob, wob, bo, (float*)d_out,
                                                      MROWS, EMBED, EMBED);
  }
}
